// MultiHeadAttentionLayer_75007308857375
// MI455X (gfx1250) — matrix operand order repaired and checked by reading, not compiled, not run on hardware
//
#include <hip/hip_runtime.h>

#define D_MODEL 1024
#define N_HEADS 16
#define HEAD    64
#define SEQ     2048
#define BATCH   2

typedef __attribute__((ext_vector_type(16))) __bf16 v16bf;
typedef __attribute__((ext_vector_type(8)))  float  v8f;
typedef unsigned int u32x4 __attribute__((ext_vector_type(4)));
typedef int          i32x8 __attribute__((ext_vector_type(8)));
typedef int          i32x4 __attribute__((ext_vector_type(4)));

union BFrag { v16bf v; unsigned u[8]; };

__device__ __forceinline__ unsigned short bf1(float f) {
  __bf16 h = (__bf16)f;                      // native v_cvt path
  return __builtin_bit_cast(unsigned short, h);
}
__device__ __forceinline__ v8f wmma_bf16(const BFrag& a, const BFrag& b, v8f c) {
  return __builtin_amdgcn_wmma_f32_16x16x32_bf16(false, a.v, false, b.v,
                                                 (short)0, c, false, false);
}

// A-fragment (16x32 bf16) from fp32 row-major source, native bf16 casts.
// Lane half lh: elems 0..15 <- K = k0+lh*16 .. +15
__device__ __forceinline__ BFrag load_a_f32(const float* __restrict__ rowp, int k0, int lh) {
  const float* ap = rowp + k0 + lh * 16;
  float4 c1 = *(const float4*)(ap);
  float4 c2 = *(const float4*)(ap + 4);
  float4 c3 = *(const float4*)(ap + 8);
  float4 c4 = *(const float4*)(ap + 12);
  BFrag a;
  a.v[0] = (__bf16)c1.x; a.v[1] = (__bf16)c1.y; a.v[2]  = (__bf16)c1.z; a.v[3]  = (__bf16)c1.w;
  a.v[4] = (__bf16)c2.x; a.v[5] = (__bf16)c2.y; a.v[6]  = (__bf16)c2.z; a.v[7]  = (__bf16)c2.w;
  a.v[8] = (__bf16)c3.x; a.v[9] = (__bf16)c3.y; a.v[10] = (__bf16)c3.z; a.v[11] = (__bf16)c3.w;
  a.v[12]= (__bf16)c4.x; a.v[13]= (__bf16)c4.y; a.v[14] = (__bf16)c4.z; a.v[15] = (__bf16)c4.w;
  return a;
}
// B-fragment (32x16 bf16) from fp32 "column source" (row of W): 16 contiguous K per lane.
__device__ __forceinline__ BFrag load_b_f32(const float* __restrict__ colp, int k0, int lh) {
  const float* bp = colp + k0 + lh * 16;
  float4 d1 = *(const float4*)(bp);
  float4 d2 = *(const float4*)(bp + 4);
  float4 d3 = *(const float4*)(bp + 8);
  float4 d4 = *(const float4*)(bp + 12);
  BFrag b;
  b.v[0] = (__bf16)d1.x; b.v[1] = (__bf16)d1.y; b.v[2]  = (__bf16)d1.z; b.v[3]  = (__bf16)d1.w;
  b.v[4] = (__bf16)d2.x; b.v[5] = (__bf16)d2.y; b.v[6]  = (__bf16)d2.z; b.v[7]  = (__bf16)d2.w;
  b.v[8] = (__bf16)d3.x; b.v[9] = (__bf16)d3.y; b.v[10] = (__bf16)d3.z; b.v[11] = (__bf16)d3.w;
  b.v[12]= (__bf16)d4.x; b.v[13]= (__bf16)d4.y; b.v[14] = (__bf16)d4.z; b.v[15] = (__bf16)d4.w;
  return b;
}
// A-fragment from bf16 source (two adjacent 16B chunks: 16 contiguous K per lane)
__device__ __forceinline__ BFrag load_a_bf16(const unsigned short* __restrict__ rowp, int k0, int lh) {
  uint4 c1 = *(const uint4*)(rowp + k0 + lh * 16);
  uint4 c2 = *(const uint4*)(rowp + k0 + lh * 16 + 8);
  BFrag a;
  a.u[0] = c1.x; a.u[1] = c1.y; a.u[2] = c1.z; a.u[3] = c1.w;
  a.u[4] = c2.x; a.u[5] = c2.y; a.u[6] = c2.z; a.u[7] = c2.w;
  return a;
}
// B-fragment from bf16 source: 16 contiguous values (32B) per lane. Works for
// global or LDS pointers (compiler keeps the address space through inlining).
__device__ __forceinline__ BFrag load_b_bf16(const unsigned short* colp, int k0, int lh) {
  const unsigned short* p = colp + k0 + lh * 16;
  uint4 c1 = *(const uint4*)(p);
  uint4 c2 = *(const uint4*)(p + 8);
  BFrag b;
  b.u[0] = c1.x; b.u[1] = c1.y; b.u[2] = c1.z; b.u[3] = c1.w;
  b.u[4] = c2.x; b.u[5] = c2.y; b.u[6] = c2.z; b.u[7] = c2.w;
  return b;
}

// ---------------- TDM: 2D tensor load -> LDS with row padding ----------------
// Copies 128 x 64 bf16 rows (contiguous, stride 64) from global to LDS,
// inserting 4 DWORDs of pad every 128B so LDS row stride = 144B (16B aligned,
// conflict-free b128 reads). D# layout per CDNA5 ISA.
#define KCHUNK_ROWS   128
#define KROW_LDS_US   72          // 64 data + 8 pad (unsigned shorts)

__device__ __forceinline__ void tdm_load_k_chunk(const unsigned short* gsrc,
                                                 unsigned lds_off) {
  unsigned long long ga = (unsigned long long)(size_t)gsrc;
  u32x4 g0;
  g0[0] = 0x1u;                                   // count=1 valid descriptor
  g0[1] = lds_off;                                // lds_addr
  g0[2] = (unsigned)(ga & 0xFFFFFFFFu);           // global_addr[31:0]
  g0[3] = (unsigned)((ga >> 32) & 0x01FFFFFFu)    // global_addr[56:32]
        | (2u << 30);                             // type = 2 ("image")
  i32x8 g1;
  g1[0] = (1 << 16)                               // data_size = 2 bytes
        | (1 << 20)                               // pad_enable
        | (4 << 22)                               // pad_interval: every 32 DWORDs (128B)
        | (3 << 25);                              // pad_amount: 4 DWORDs
  g1[1] = (int)((HEAD & 0xFFFFu) << 16);          // tensor_dim0[15:0] = 64
  g1[2] = (int)(((unsigned)KCHUNK_ROWS & 0xFFFFu) << 16); // dim0 hi=0 | tensor_dim1 lo = 128
  g1[3] = (int)((HEAD & 0xFFFFu) << 16);          // tensor_dim1 hi=0 | tile_dim0 = 64
  g1[4] = (int)(KCHUNK_ROWS & 0xFFFFu);           // tile_dim1 = 128, tile_dim2 = 0
  g1[5] = HEAD;                                   // tensor_dim0_stride = 64
  g1[6] = 0;                                      // stride0 hi | stride1 lo
  g1[7] = 0;                                      // stride1 hi
  i32x4 z4 = {0, 0, 0, 0};
  i32x8 z8 = {0, 0, 0, 0, 0, 0, 0, 0};
  // 6-arg toolchain signature: (g0, g1, g2, g3, extra, cpol)
  __builtin_amdgcn_tensor_load_to_lds(g0, g1, z4, z4, z8, 0);
}

// -------- Kernel 1: QKV projection + bias + (center,L2-normalize for Q,K) --------
// grid(64, 16, 3), block 128 (4 waves). Wave -> one 16(M)x64(head) tile.
__global__ void __launch_bounds__(128) qkv_proj_norm_kernel(
    const float* __restrict__ qin, const float* __restrict__ kin, const float* __restrict__ vin,
    const float* __restrict__ Wq, const float* __restrict__ bq,
    const float* __restrict__ Wk, const float* __restrict__ bk,
    const float* __restrict__ Wv, const float* __restrict__ bv,
    unsigned short* __restrict__ Qn, unsigned short* __restrict__ Kn,
    unsigned short* __restrict__ Vt) {
  const int lane = threadIdx.x & 31;
  const int wv = threadIdx.x >> 5;
  const int lh = lane >> 4, l15 = lane & 15;
  const int which = blockIdx.z;
  const float* X    = (which == 0) ? qin : (which == 1) ? kin : vin;
  const float* W    = (which == 0) ? Wq  : (which == 1) ? Wk  : Wv;
  const float* bias = (which == 0) ? bq  : (which == 1) ? bk  : bv;
  const int h = blockIdx.y;
  const int m = (blockIdx.x * 4 + wv) * 16;
  const float* arow = X + (size_t)(m + l15) * D_MODEL;

  v8f acc[4] = {};
  for (int k0 = 0; k0 < D_MODEL; k0 += 32) {
    __builtin_prefetch(arow + k0 + 128, 0, 0);
    BFrag a = load_a_f32(arow, k0, lh);
#pragma unroll
    for (int t = 0; t < 4; ++t) {
      const float* wrow = W + (size_t)(h * HEAD + t * 16 + l15) * D_MODEL;
      BFrag bf = load_b_f32(wrow, k0, lh);
      acc[t] = wmma_bf16(a, bf, acc[t]);
    }
  }
#pragma unroll
  for (int t = 0; t < 4; ++t) {
    float bval = bias[h * HEAD + t * 16 + l15];
#pragma unroll
    for (int r = 0; r < 8; ++r) acc[t][r] += bval;
  }

  if (which < 2) {
    float s[8];
#pragma unroll
    for (int r = 0; r < 8; ++r) s[r] = acc[0][r] + acc[1][r] + acc[2][r] + acc[3][r];
#pragma unroll
    for (int r = 0; r < 8; ++r) {
#pragma unroll
      for (int off = 1; off < 16; off <<= 1) s[r] += __shfl_xor(s[r], off, 16);
    }
#pragma unroll
    for (int r = 0; r < 8; ++r) {
      float mean = s[r] * (1.0f / 64.0f);
#pragma unroll
      for (int t = 0; t < 4; ++t) acc[t][r] -= mean;
    }
    float sq[8];
#pragma unroll
    for (int r = 0; r < 8; ++r) {
      sq[r] = acc[0][r]*acc[0][r] + acc[1][r]*acc[1][r] +
              acc[2][r]*acc[2][r] + acc[3][r]*acc[3][r];
#pragma unroll
      for (int off = 1; off < 16; off <<= 1) sq[r] += __shfl_xor(sq[r], off, 16);
    }
#pragma unroll
    for (int r = 0; r < 8; ++r) {
      float inv = 1.0f / fmaxf(sqrtf(sq[r]), 1e-12f);
#pragma unroll
      for (int t = 0; t < 4; ++t) acc[t][r] *= inv;
    }
    unsigned short* out = (which == 0) ? Qn : Kn;
#pragma unroll
    for (int t = 0; t < 4; ++t) {
#pragma unroll
      for (int r = 0; r < 8; ++r) {
        int rg = m + r + 8 * lh;
        int bb = rg >> 11, srow = rg & (SEQ - 1);
        size_t idx = (((size_t)(bb * N_HEADS + h) * SEQ) + srow) * HEAD + t * 16 + l15;
        out[idx] = bf1(acc[t][r]);
      }
    }
  } else {
    // V stored transposed: Vt[b][h][col][s]
#pragma unroll
    for (int t = 0; t < 4; ++t) {
#pragma unroll
      for (int r = 0; r < 8; ++r) {
        int rg = m + r + 8 * lh;
        int bb = rg >> 11, srow = rg & (SEQ - 1);
        int col = t * 16 + l15;
        size_t idx = (((size_t)(bb * N_HEADS + h) * HEAD) + col) * SEQ + srow;
        Vt[idx] = bf1(acc[t][r]);
      }
    }
  }
}

// -------- Kernel 2: scores = scale*Q.K^T, mask, softmax -> attn (fp32, to d_out) --------
// grid(128, 16, 2), block 256 (8 waves). Block -> 16 query rows x all 2048 keys.
// K rows are fed by the Tensor Data Mover into a double-buffered, row-padded
// LDS staging buffer (wave 0 drives the TDM; completion via s_wait_tensorcnt).
__global__ void __launch_bounds__(256) score_softmax_kernel(
    const unsigned short* __restrict__ Qn, const unsigned short* __restrict__ Kn,
    const int* __restrict__ mask, const float* __restrict__ scale_p,
    float* __restrict__ attn) {
  __shared__ __align__(16) unsigned short sK[2][KCHUNK_ROWS * KROW_LDS_US];
  __shared__ float redm[8][16];
  __shared__ float redsum[8][16];

  const int lane = threadIdx.x & 31;
  const int wv = threadIdx.x >> 5;         // 0..7
  const int lh = lane >> 4, l15 = lane & 15;
  const int mt = blockIdx.x;               // 0..127
  const int h = blockIdx.y;
  const int b = blockIdx.z;
  const float scale = scale_p[0];

  const size_t headBase = (size_t)(b * N_HEADS + h) * SEQ * HEAD;
  const unsigned short* kbase = Kn + headBase;
  const unsigned lds0 = (unsigned)(size_t)&sK[0][0];
  const unsigned lds1 = (unsigned)(size_t)&sK[1][0];

  // Q fragments for this 16-row tile (K-dim = 64 -> 2 steps of 32)
  BFrag aq[2];
  {
    const unsigned short* qr = Qn + headBase + (size_t)(mt * 16 + l15) * HEAD;
    aq[0] = load_a_bf16(qr, 0, lh);
    aq[1] = load_a_bf16(qr, 32, lh);
  }

  // prologue: TDM chunk 0 -> buffer 0
  if (wv == 0) tdm_load_k_chunk(kbase, lds0);

  v8f acc[16] = {};
#pragma unroll
  for (int c = 0; c < 16; ++c) {           // 16 chunks of 128 key rows
    if (wv == 0) {
      if (c + 1 < 16) {
        tdm_load_k_chunk(kbase + (size_t)(c + 1) * KCHUNK_ROWS * HEAD,
                         ((c + 1) & 1) ? lds1 : lds0);
        __builtin_amdgcn_s_wait_tensorcnt(1);   // oldest (chunk c) complete
      } else {
        __builtin_amdgcn_s_wait_tensorcnt(0);
      }
    }
    __syncthreads();                       // chunk c visible to all waves
    // wave wv consumes key tile nt = c*8 + wv -> LDS rows wv*16 .. wv*16+15
    const unsigned short* kr = &sK[c & 1][(wv * 16 + l15) * KROW_LDS_US];
    BFrag b0 = load_b_bf16(kr, 0, lh);
    BFrag b1 = load_b_bf16(kr, 32, lh);
    acc[c] = wmma_bf16(aq[0], b0, acc[c]);
    acc[c] = wmma_bf16(aq[1], b1, acc[c]);
    __syncthreads();                       // done reading before buffer reuse
  }

  // scale + mask, track row max
  float pmax[8];
#pragma unroll
  for (int r = 0; r < 8; ++r) pmax[r] = -3.0e38f;
#pragma unroll
  for (int i = 0; i < 16; ++i) {
    int col = (i * 8 + wv) * 16 + l15;
    int mv = mask[b * SEQ + col];
#pragma unroll
    for (int r = 0; r < 8; ++r) {
      float sc = (mv == 0) ? -1.0e9f : acc[i][r] * scale;
      acc[i][r] = sc;
      pmax[r] = fmaxf(pmax[r], sc);
    }
  }
#pragma unroll
  for (int r = 0; r < 8; ++r) {
#pragma unroll
    for (int off = 1; off < 16; off <<= 1)
      pmax[r] = fmaxf(pmax[r], __shfl_xor(pmax[r], off, 16));
  }
  if (l15 == 0) {                          // lanes 0 and 16
#pragma unroll
    for (int r = 0; r < 8; ++r) redm[wv][lh * 8 + r] = pmax[r];
  }
  __syncthreads();
  float rm[8];
#pragma unroll
  for (int r = 0; r < 8; ++r) {
    float m0 = redm[0][lh * 8 + r];
#pragma unroll
    for (int w = 1; w < 8; ++w) m0 = fmaxf(m0, redm[w][lh * 8 + r]);
    rm[r] = m0;
  }
  // exp + row sum
  float esum[8];
#pragma unroll
  for (int r = 0; r < 8; ++r) esum[r] = 0.0f;
#pragma unroll
  for (int i = 0; i < 16; ++i) {
#pragma unroll
    for (int r = 0; r < 8; ++r) {
      float e = __expf(acc[i][r] - rm[r]);
      acc[i][r] = e;
      esum[r] += e;
    }
  }
#pragma unroll
  for (int r = 0; r < 8; ++r) {
#pragma unroll
    for (int off = 1; off < 16; off <<= 1) esum[r] += __shfl_xor(esum[r], off, 16);
  }
  if (l15 == 0) {
#pragma unroll
    for (int r = 0; r < 8; ++r) redsum[wv][lh * 8 + r] = esum[r];
  }
  __syncthreads();
  float inv[8];
#pragma unroll
  for (int r = 0; r < 8; ++r) {
    float s0 = redsum[0][lh * 8 + r];
#pragma unroll
    for (int w = 1; w < 8; ++w) s0 += redsum[w][lh * 8 + r];
    inv[r] = 1.0f / s0;
  }
  // write attn
  const size_t attnBase = (size_t)(b * N_HEADS + h) * SEQ * SEQ + (size_t)(mt * 16) * SEQ;
#pragma unroll
  for (int i = 0; i < 16; ++i) {
    int col = (i * 8 + wv) * 16 + l15;
#pragma unroll
    for (int r = 0; r < 8; ++r) {
      attn[attnBase + (size_t)(r + 8 * lh) * SEQ + col] = acc[i][r] * inv[r];
    }
  }
}

// -------- Kernel 3: ctx_head = attn @ V  (reads fp32 attn from d_out, bf16 WMMA) --------
// grid(32, 16, 2), block 128 (4 waves). Wave -> 16(M)x64(head) tile, K-loop over 2048.
__global__ void __launch_bounds__(128) pv_kernel(
    const float* __restrict__ attn, const unsigned short* __restrict__ Vt,
    float* __restrict__ ctxh) {
  const int lane = threadIdx.x & 31;
  const int wv = threadIdx.x >> 5;
  const int lh = lane >> 4, l15 = lane & 15;
  const int mt = blockIdx.x * 4 + wv;      // 0..127
  const int h = blockIdx.y;
  const int b = blockIdx.z;

  const float* arow = attn + (size_t)(b * N_HEADS + h) * SEQ * SEQ
                           + (size_t)(mt * 16 + l15) * SEQ;
  const size_t vtBase = (size_t)(b * N_HEADS + h) * HEAD * SEQ;

  v8f acc[4] = {};
  for (int k0 = 0; k0 < SEQ; k0 += 32) {
    __builtin_prefetch(arow + k0 + 128, 0, 0);
    BFrag a = load_a_f32(arow, k0, lh);
#pragma unroll
    for (int t = 0; t < 4; ++t) {
      const unsigned short* vcol = Vt + vtBase + (size_t)(t * 16 + l15) * SEQ;
      BFrag bf = load_b_bf16(vcol, k0, lh);
      acc[t] = wmma_bf16(a, bf, acc[t]);
    }
  }
#pragma unroll
  for (int t = 0; t < 4; ++t) {
#pragma unroll
    for (int r = 0; r < 8; ++r) {
      int srow = mt * 16 + r + 8 * lh;
      size_t idx = ((size_t)(b * SEQ + srow)) * D_MODEL + h * HEAD + t * 16 + l15;
      ctxh[idx] = acc[t][r];
    }
  }
}

// -------- Kernel 4: ctx = ctxh @ Wo^T + bo  -> d_out (fp32) --------
__global__ void __launch_bounds__(128) oproj_kernel(
    const float* __restrict__ ctxh, const float* __restrict__ Wo,
    const float* __restrict__ bo, float* __restrict__ out) {
  const int lane = threadIdx.x & 31;
  const int wv = threadIdx.x >> 5;
  const int lh = lane >> 4, l15 = lane & 15;
  const int m = (blockIdx.x * 4 + wv) * 16;
  const int g = blockIdx.y;
  const float* arow = ctxh + (size_t)(m + l15) * D_MODEL;

  v8f acc[4] = {};
  for (int k0 = 0; k0 < D_MODEL; k0 += 32) {
    __builtin_prefetch(arow + k0 + 128, 0, 0);
    BFrag a = load_a_f32(arow, k0, lh);
#pragma unroll
    for (int t = 0; t < 4; ++t) {
      const float* wrow = Wo + (size_t)(g * 64 + t * 16 + l15) * D_MODEL;
      BFrag bf = load_b_f32(wrow, k0, lh);
      acc[t] = wmma_bf16(a, bf, acc[t]);
    }
  }
#pragma unroll
  for (int t = 0; t < 4; ++t) {
    float bval = bo[g * 64 + t * 16 + l15];
#pragma unroll
    for (int r = 0; r < 8; ++r) {
      int rg = m + r + 8 * lh;
      out[(size_t)rg * D_MODEL + g * 64 + t * 16 + l15] = acc[t][r] + bval;
    }
  }
}

extern "C" void kernel_launch(void* const* d_in, const int* in_sizes, int n_in,
                              void* d_out, int out_size, void* d_ws, size_t ws_size,
                              hipStream_t stream) {
  (void)in_sizes; (void)n_in; (void)out_size; (void)ws_size;
  const float* qin  = (const float*)d_in[0];
  const float* kin  = (const float*)d_in[1];
  const float* vin  = (const float*)d_in[2];
  const int*   mask = (const int*)  d_in[3];
  const float* Wq   = (const float*)d_in[4];
  const float* bq   = (const float*)d_in[5];
  const float* Wk   = (const float*)d_in[6];
  const float* bk   = (const float*)d_in[7];
  const float* Wv   = (const float*)d_in[8];
  const float* bv   = (const float*)d_in[9];
  const float* Wo   = (const float*)d_in[10];
  const float* bo   = (const float*)d_in[11];
  const float* scale= (const float*)d_in[12];

  float* ctx_out  = (float*)d_out;                                  // [2,2048,1024]
  float* attn_out = ctx_out + (size_t)BATCH * SEQ * D_MODEL;        // [2,16,2048,2048]

  const size_t NE = (size_t)BATCH * N_HEADS * SEQ * HEAD;           // 4M elems
  unsigned short* Qn  = (unsigned short*)d_ws;                      // bf16 [B,H,S,64]
  unsigned short* Kn  = Qn + NE;                                    // bf16 [B,H,S,64]
  unsigned short* Vt  = Kn + NE;                                    // bf16 [B,H,64,S]
  float*          ctxh = (float*)(Vt + NE);                         // fp32 [B,S,1024]

  qkv_proj_norm_kernel<<<dim3(64, 16, 3), 128, 0, stream>>>(
      qin, kin, vin, Wq, bq, Wk, bk, Wv, bv, Qn, Kn, Vt);
  score_softmax_kernel<<<dim3(128, 16, 2), 256, 0, stream>>>(
      Qn, Kn, mask, scale, attn_out);
  pv_kernel<<<dim3(32, 16, 2), 128, 0, stream>>>(attn_out, Vt, ctxh);
  oproj_kernel<<<dim3(64, 16, 1), 128, 0, stream>>>(ctxh, Wo, bo, ctx_out);
}
